// EGNNA_22179211117031
// MI455X (gfx1250) — matrix operand order repaired and checked by reading, not compiled, not run on hardware
//
#include <hip/hip_runtime.h>
#include <hip/hip_bf16.h>

// ---------------------------------------------------------------------------
// EGNNA pipeline for MI455X (gfx1250, wave32, WMMA).
// Encode: 3 GAT layers, fp32 GEMMs on v_wmma_f32_16x16x4_f32 (fp32-exact
// matrix pipe, B pre-transposed so every fragment is one b64 load, K-block
// staged loads so WMMAs pipeline instead of serializing on loadcnt).
// Decode: factored pairwise MLP, layer-2 GEMM (589824x256x128, ~39 GFLOP)
// on v_wmma_f32_16x16x32_bf16 with in-register A-fragment construction.
// ---------------------------------------------------------------------------

typedef __attribute__((ext_vector_type(16))) __bf16 v16bf;
typedef __attribute__((ext_vector_type(8)))  float  v8f;
typedef __attribute__((ext_vector_type(2)))  float  v2f;

#define NNODES 768
#define ECH 4

// ---------------- tiled transpose: dst[c][r] = src[r][c] -------------------
__global__ void egnna_transpose(const float* __restrict__ src,
                                float* __restrict__ dst, int R, int Cc) {
    __shared__ float tile[32][33];
    const int c0 = blockIdx.x * 32, r0 = blockIdx.y * 32;
    const int tx = threadIdx.x, ty = threadIdx.y;  // 32 x 8
#pragma unroll
    for (int dy = 0; dy < 32; dy += 8)
        tile[ty + dy][tx] = src[(size_t)(r0 + ty + dy) * Cc + (c0 + tx)];
    __syncthreads();
#pragma unroll
    for (int dy = 0; dy < 32; dy += 8)
        dst[(size_t)(c0 + ty + dy) * R + (r0 + tx)] = tile[tx][ty + dy];
}

// ---------------- fp32 WMMA GEMM: C = scale*(A@BT^T) [+ bias] [+ C] --------
// One wave computes a 16x64 strip; K stepped 16 at a time (4 WMMAs deep),
// loads staged into distinct registers so the scheduler can clause + overlap.
// A-frag (16x4): lane L -> row L&15, K-pair {2*hi, 2*hi+1}  (hi = L>>4)
// B-frag (4x16): lane L -> col n0+nt*16+(L&15), rows {k+2*hi, k+2*hi+1}
//                (contiguous in BT[N][K] -> single b64 load)
// C-frag: VGPR v -> (row = v + 8*hi, col = n0+nt*16+(L&15))
__global__ __launch_bounds__(32) void egnna_gemm_wmma_f32(
    const float* __restrict__ A, const float* __restrict__ BT,
    const float* __restrict__ bias, float* __restrict__ C,
    int M, int N, int K, float scale, int accf) {
    const int lane = threadIdx.x & 31;
    const int m  = lane & 15;
    const int hi = lane >> 4;
    const int row = blockIdx.y * 16 + m;
    const int n0  = blockIdx.x * 64;

    v8f acc[4];
#pragma unroll
    for (int nt = 0; nt < 4; ++nt) acc[nt] = (v8f){0.f,0.f,0.f,0.f,0.f,0.f,0.f,0.f};

    const float* Arow = A + (size_t)row * K + 2 * hi;
    const float* Bc0 = BT + (size_t)(n0 +  0 + m) * K + 2 * hi;
    const float* Bc1 = BT + (size_t)(n0 + 16 + m) * K + 2 * hi;
    const float* Bc2 = BT + (size_t)(n0 + 32 + m) * K + 2 * hi;
    const float* Bc3 = BT + (size_t)(n0 + 48 + m) * K + 2 * hi;

    for (int kk = 0; kk < K; kk += 16) {   // 4 WMMA K-steps per block
        v2f a[4], b[4][4];
#pragma unroll
        for (int u = 0; u < 4; ++u) {
            a[u]    = *reinterpret_cast<const v2f*>(Arow + kk + 4 * u);
            b[u][0] = *reinterpret_cast<const v2f*>(Bc0  + kk + 4 * u);
            b[u][1] = *reinterpret_cast<const v2f*>(Bc1  + kk + 4 * u);
            b[u][2] = *reinterpret_cast<const v2f*>(Bc2  + kk + 4 * u);
            b[u][3] = *reinterpret_cast<const v2f*>(Bc3  + kk + 4 * u);
        }
#pragma unroll
        for (int u = 0; u < 4; ++u)
#pragma unroll
            for (int nt = 0; nt < 4; ++nt)
                acc[nt] = __builtin_amdgcn_wmma_f32_16x16x4_f32(
                    false, a[u], false, b[u][nt], (short)0, acc[nt], false, false);
    }

    const int r0 = blockIdx.y * 16 + 8 * hi;
#pragma unroll
    for (int nt = 0; nt < 4; ++nt) {
        const int col = n0 + nt * 16 + m;
        const float bv = bias ? bias[col] : 0.f;
#pragma unroll
        for (int v = 0; v < 8; ++v) {
            const size_t idx = (size_t)(r0 + v) * N + col;
            float val = scale * acc[nt][v] + bv;
            if (accf) val += C[idx];
            C[idx] = val;
        }
    }
}

// ---------------- attention features: fsrc/fdst [N,E] ----------------------
__global__ void egnna_attn_feats(const float* __restrict__ h,
                                 const float* __restrict__ a,
                                 float* __restrict__ fsrc,
                                 float* __restrict__ fdst,
                                 int N, int D) {
    int idx = blockIdx.x * blockDim.x + threadIdx.x;  // n*ECH + e
    if (idx >= N * ECH) return;
    int n = idx / ECH, e = idx % ECH;
    const float* hr = h + n * D;
    const float* as = a + e * 2 * D;
    float s0 = 0.f, s1 = 0.f;
    for (int k = 0; k < D; ++k) {
        s0 += hr[k] * as[k];
        s1 += hr[k] * as[D + k];
    }
    fsrc[idx] = s0;
    fdst[idx] = s1;
}

// ---------------- masked softmax row (one channel) -------------------------
__global__ void egnna_alpha_row(const float* __restrict__ adj,
                                const float* __restrict__ fsrc,
                                const float* __restrict__ fdst,
                                float* __restrict__ alpha,
                                int e) {
    const int N = NNODES;
    int n = blockIdx.x;
    const float* adjrow = adj + ((size_t)e * N + n) * N;
    float fs = fsrc[n * ECH + e];
    __shared__ float red[256];
    float vals[3];
    float mx = -2e9f;
#pragma unroll
    for (int t = 0; t < 3; ++t) {
        int m = threadIdx.x + t * 256;
        float v;
        if (adjrow[m] > 0.87f) {
            float x = fs + fdst[m * ECH + e];
            v = x > 0.f ? x : 0.2f * x;          // leaky_relu(0.2)
        } else {
            v = -1e9f;
        }
        vals[t] = v;
        mx = fmaxf(mx, v);
    }
    red[threadIdx.x] = mx;
    __syncthreads();
    for (int s = 128; s > 0; s >>= 1) {
        if (threadIdx.x < s) red[threadIdx.x] = fmaxf(red[threadIdx.x], red[threadIdx.x + s]);
        __syncthreads();
    }
    mx = red[0];
    __syncthreads();
    float lsum = 0.f;
#pragma unroll
    for (int t = 0; t < 3; ++t) {
        vals[t] = __expf(vals[t] - mx);
        lsum += vals[t];
    }
    red[threadIdx.x] = lsum;
    __syncthreads();
    for (int s = 128; s > 0; s >>= 1) {
        if (threadIdx.x < s) red[threadIdx.x] += red[threadIdx.x + s];
        __syncthreads();
    }
    float inv = 1.0f / red[0];
#pragma unroll
    for (int t = 0; t < 3; ++t) {
        int m = threadIdx.x + t * 256;
        alpha[(size_t)n * N + m] = vals[t] * inv;
    }
}

// ---------------- elementwise helpers --------------------------------------
__global__ void egnna_relu(float* x, int n) {
    int i = blockIdx.x * blockDim.x + threadIdx.x;
    if (i < n) x[i] = fmaxf(x[i], 0.f);
}
__global__ void egnna_add(float* dst, const float* src, int n) {
    int i = blockIdx.x * blockDim.x + threadIdx.x;
    if (i < n) dst[i] += src[i];
}
__global__ void egnna_norms(const float* __restrict__ h, float* __restrict__ norms,
                            int N, int D) {
    int i = blockIdx.x * blockDim.x + threadIdx.x;
    if (i >= N) return;
    float s = 0.f;
    for (int k = 0; k < D; ++k) { float v = h[i * D + k]; s += v * v; }
    norms[i] = sqrtf(s);
}

// ---------------- pack fc2_w [256,128] fp32 -> WMMA B-fragment bf16 --------
// B (32x16 per tile): lanes 0-15 hold K=0..15 (half idx = k), lanes 16-31
// hold K=16..31; column N = lane&15. Tiles ordered (kt major, nt minor).
__global__ void egnna_packB(const float* __restrict__ W, __hip_bfloat16* __restrict__ Bp) {
    int idx = blockIdx.x * blockDim.x + threadIdx.x;  // over 256*128
    if (idx >= 256 * 128) return;
    int k = idx / 128, n = idx % 128;
    int kt = k >> 5, kl = k & 31;
    int nt = n >> 4;
    int lane = ((kl < 16) ? 0 : 16) + (n & 15);
    int hh = kl & 15;
    Bp[(((kt * 8 + nt) * 32 + lane) * 16) + hh] = __float2bfloat16(W[idx]);
}

// ---------------- fused decode: one wave = 16 pairs (fixed i, 16 j's) ------
__global__ __launch_bounds__(256) void egnna_decode(
    const float* __restrict__ P,        // [768,256] = h@fc1a + b1
    const float* __restrict__ Q,        // [768,256] = h@fc1b
    const float* __restrict__ R,        // [4,256]   = fc1_w rows 256..259
    const float* __restrict__ raw_adj,  // [4,768,768]
    const __hip_bfloat16* __restrict__ Bp,  // packed fc2_w
    const float* __restrict__ fc2_b,    // [128]
    const float* __restrict__ fc3_w,    // [128,2]
    const float* __restrict__ fc3_b,    // [2]
    const float* __restrict__ hmat,     // [768,128]
    const float* __restrict__ norms,    // [768]
    float* __restrict__ out,            // [768*768,2]
    float* __restrict__ cosout)         // [768*768]
{
    const int N = NNODES;
    const size_t NN = (size_t)N * N;
    const int wave = threadIdx.x >> 5;
    const int lane = threadIdx.x & 31;
    const int i  = blockIdx.x / 6;
    const int jg = blockIdx.x % 6;
    const int j0 = (jg * 8 + wave) * 16;
    const int m  = lane & 15;   // A-row / B-column slot
    const int hi = lane >> 4;   // half-wave select
    const int j  = j0 + m;

    // per-lane edge-feature scalars for pair (i, j)
    const size_t pij = (size_t)i * N + j;
    const float r0 = raw_adj[0 * NN + pij];
    const float r1 = raw_adj[1 * NN + pij];
    const float r2 = raw_adj[2 * NN + pij];
    const float r3 = raw_adj[3 * NN + pij];

    const float* Pi = P + i * 256;
    const float* Qj = Q + j * 256;

    v8f acc[8];
#pragma unroll
    for (int nt = 0; nt < 8; ++nt) acc[nt] = (v8f){0.f, 0.f, 0.f, 0.f, 0.f, 0.f, 0.f, 0.f};

    // layer-2 GEMM: hdn1(16x256, built in-register) @ fc2(256x128)
#pragma unroll
    for (int kt = 0; kt < 8; ++kt) {
        v16bf a;
        const int kb = kt * 32 + hi * 16;
#pragma unroll
        for (int ch = 0; ch < 2; ++ch) {
            const int k0 = kb + ch * 8;
#pragma unroll
            for (int t = 0; t < 8; ++t) {
                const int k = k0 + t;
                float v = Pi[k] + Qj[k] + r0 * R[k] + r1 * R[256 + k] +
                          r2 * R[512 + k] + r3 * R[768 + k];
                v = v > 0.f ? v : 0.f;           // relu(hdn1)
                a[ch * 8 + t] = (__bf16)v;
            }
        }
#pragma unroll
        for (int nt = 0; nt < 8; ++nt) {
            v16bf b = *reinterpret_cast<const v16bf*>(Bp + (((kt * 8 + nt) * 32 + lane) * 16));
            acc[nt] = __builtin_amdgcn_wmma_f32_16x16x32_bf16(
                false, a, false, b, (short)0, acc[nt], false, false);
        }
    }

    // layer 3: per-lane partial logits over its 8 columns, rows M = hi*8 + v
    float pl[8][2];
#pragma unroll
    for (int v = 0; v < 8; ++v) { pl[v][0] = 0.f; pl[v][1] = 0.f; }
#pragma unroll
    for (int nt = 0; nt < 8; ++nt) {
        const int n = nt * 16 + m;
        const float bias = fc2_b[n];
        const float w0 = fc3_w[n * 2 + 0];
        const float w1 = fc3_w[n * 2 + 1];
#pragma unroll
        for (int v = 0; v < 8; ++v) {
            float x = acc[nt][v] + bias;
            x = x > 0.f ? x : 0.f;               // relu(hdn2)
            pl[v][0] += x * w0;
            pl[v][1] += x * w1;
        }
    }
    // reduce partials across the 16 lanes of each half-wave
#pragma unroll
    for (int v = 0; v < 8; ++v) {
#pragma unroll
        for (int c = 0; c < 2; ++c) {
            float s = pl[v][c];
            s += __shfl_xor(s, 1, 32);
            s += __shfl_xor(s, 2, 32);
            s += __shfl_xor(s, 4, 32);
            s += __shfl_xor(s, 8, 32);
            pl[v][c] = s;
        }
    }

    // cosine similarity for pair (i, j0+m): split the 128-dot across halves
    const float* hri = hmat + i * 128;
    const float* hrj = hmat + (j0 + m) * 128;
    float dotp = 0.f;
#pragma unroll 4
    for (int k = hi * 64; k < hi * 64 + 64; ++k) dotp += hri[k] * hrj[k];
    dotp += __shfl_xor(dotp, 16, 32);

    if (m < 8) {  // lanes hi*16 + r (r<8) write row M = hi*8 + r
        const int row = hi * 8 + m;
        const size_t p = (size_t)i * N + (j0 + row);
        float l0 = pl[m][0] + fc3_b[0];
        float l1 = pl[m][1] + fc3_b[1];
        float mx = fmaxf(l0, l1);
        float lse = mx + __logf(__expf(l0 - mx) + __expf(l1 - mx));
        out[p * 2 + 0] = l0 - lse;
        out[p * 2 + 1] = l1 - lse;
    }
    if (hi == 0) {
        const size_t p = (size_t)i * N + (j0 + m);
        cosout[p] = dotp / fmaxf(norms[i] * norms[j0 + m], 1e-6f);
    }
}

// ---------------------------------------------------------------------------
extern "C" void kernel_launch(void* const* d_in, const int* in_sizes, int n_in,
                              void* d_out, int out_size, void* d_ws, size_t ws_size,
                              hipStream_t stream) {
    const float* x       = (const float*)d_in[0];
    const float* adj     = (const float*)d_in[1];
    const float* raw_adj = (const float*)d_in[2];
    const float* W1      = (const float*)d_in[3];
    const float* a1      = (const float*)d_in[4];
    const float* W2      = (const float*)d_in[5];
    const float* a2      = (const float*)d_in[6];
    const float* fc1_w   = (const float*)d_in[7];
    const float* fc1_b   = (const float*)d_in[8];
    const float* fc2_w   = (const float*)d_in[9];
    const float* fc2_b   = (const float*)d_in[10];
    const float* fc3_w   = (const float*)d_in[11];
    const float* fc3_b   = (const float*)d_in[12];
    float* outp = (float*)d_out;

    const int N = NNODES;
    float* w = (float*)d_ws;
    // workspace layout (floats)
    float* hbuf  = w;                       // 768*256
    float* ybuf  = hbuf + 196608;           // 768*256
    float* zbuf  = ybuf + 196608;           // 768*256 (becomes y+z)
    float* h3    = zbuf + 196608;           // 768*128
    float* fsrc  = h3 + 98304;              // 768*4
    float* fdst  = fsrc + 3072;             // 768*4
    float* albuf = fdst + 3072;             // 768*768
    float* Pbuf  = albuf + 589824;          // 768*256
    float* Qbuf  = Pbuf + 196608;           // 768*256
    float* nrm   = Qbuf + 196608;           // 768
    float* BTbuf = nrm + 768;               // up to 768*256 (transposed B)
    __hip_bfloat16* Bp = (__hip_bfloat16*)(BTbuf + 196608);  // 256*128 bf16

    // C = scale*(A@B) [+bias] [+C]   via transpose + fp32 WMMA
    auto gemm = [&](const float* A, const float* B, const float* bias, float* C,
                    int M, int Nn, int K, float scale, int accf) {
        egnna_transpose<<<dim3(Nn / 32, K / 32), dim3(32, 8), 0, stream>>>(B, BTbuf, K, Nn);
        egnna_gemm_wmma_f32<<<dim3(Nn / 64, M / 16), 32, 0, stream>>>(
            A, BTbuf, bias, C, M, Nn, K, scale, accf);
    };

    auto gat = [&](const float* hin, const float* W, const float* a,
                   int Din, int Dout, float* outbuf) {
        gemm(hin, W, nullptr, hbuf, N, Dout, Din, 1.f, 0);        // h = hin @ W
        egnna_attn_feats<<<(N * ECH + 255) / 256, 256, 0, stream>>>(
            hbuf, a, fsrc, fdst, N, Dout);
        // transpose h once; reused by all 4 channel aggregations
        egnna_transpose<<<dim3(Dout / 32, N / 32), dim3(32, 8), 0, stream>>>(
            hbuf, BTbuf, N, Dout);
        for (int e = 0; e < ECH; ++e) {
            egnna_alpha_row<<<N, 256, 0, stream>>>(adj, fsrc, fdst, albuf, e);
            egnna_gemm_wmma_f32<<<dim3(Dout / 64, N / 16), 32, 0, stream>>>(
                albuf, BTbuf, nullptr, outbuf, N, Dout, N, 0.25f, e > 0 ? 1 : 0);
        }
    };

    // encode
    gat(x, W1, a1, 256, 256, ybuf);
    egnna_relu<<<(196608 + 255) / 256, 256, 0, stream>>>(ybuf, 196608);
    gat(ybuf, W1, a1, 256, 256, zbuf);
    egnna_add<<<(196608 + 255) / 256, 256, 0, stream>>>(zbuf, ybuf, 196608);
    gat(zbuf, W2, a2, 256, 128, h3);

    // decode precompute
    egnna_norms<<<(N + 255) / 256, 256, 0, stream>>>(h3, nrm, N, 128);
    gemm(h3, fc1_w, fc1_b, Pbuf, N, 256, 128, 1.f, 0);               // P = h@fc1[:128] + b1
    gemm(h3, fc1_w + 128 * 256, nullptr, Qbuf, N, 256, 128, 1.f, 0); // Q = h@fc1[128:256]
    egnna_packB<<<(256 * 128 + 255) / 256, 256, 0, stream>>>(fc2_w, Bp);

    // fused pairwise decode (WMMA bf16): 768 i-rows x 6 j-groups, 8 waves/WG
    egnna_decode<<<N * 6, 256, 0, stream>>>(
        Pbuf, Qbuf, fc1_w + 256 * 256, raw_adj, Bp, fc2_b, fc3_w, fc3_b,
        h3, nrm, outp, outp + (size_t)2 * N * N);
}
